// PatchSpatialAttention_16363825398260
// MI455X (gfx1250) — matrix operand order repaired and checked by reading, not compiled, not run on hardware
//
#include <hip/hip_runtime.h>
#include <hip/hip_bf16.h>

typedef __attribute__((ext_vector_type(16))) _Float16 v16h;
typedef __attribute__((ext_vector_type(8)))  _Float16 v8h;
typedef __attribute__((ext_vector_type(8)))  float    v8f;
typedef __attribute__((ext_vector_type(4)))  float    v4f;

#define B_   2
#define C_   1024
#define H_   256
#define W_   256
#define HF_  32
#define WF_  32
#define P_   1024          // HF_*WF_
#define PPAD 1026          // one zero row of padding at each end of the patch axis

// workspace layout (byte offsets)
#define G_OFF   0u                     // float   g[B][C][P]      8 MiB
#define YT_OFF  (8u * 1024 * 1024)     // f16     yt[B][PPAD][C]  ~4 MiB
#define WT_OFF  (16u * 1024 * 1024)    // f16     wt[3][C][C]     6 MiB

// ---------------------------------------------------------------------------
// Kernel 1: convert conv_w [co][ci][3] f32  ->  wt[tap][co][ci] f16
// ---------------------------------------------------------------------------
__global__ void wcvt_kernel(const float* __restrict__ w, _Float16* __restrict__ wt) {
    int tid = blockIdx.x * blockDim.x + threadIdx.x;   // 0 .. C_*C_-1
    int co = tid >> 10;
    int ci = tid & 1023;
    const float* p = w + ((size_t)co * (C_ * 3) + (size_t)ci * 3);
    float w0 = p[0], w1 = p[1], w2 = p[2];
    size_t o = (size_t)co * C_ + ci;
    wt[0u * C_ * C_ + o] = (_Float16)w0;
    wt[1u * C_ * C_ + o] = (_Float16)w1;
    wt[2u * C_ * C_ + o] = (_Float16)w2;
}

// ---------------------------------------------------------------------------
// Kernel 2: 8x8 patch mean pool, write transposed f16: yt[b][p+1][ci]
// One block per (b, c, ph): 256 threads = 256 columns of one 8-row slab.
// ---------------------------------------------------------------------------
__global__ void pool_kernel(const float* __restrict__ x, _Float16* __restrict__ yt) {
    int bid = blockIdx.x;
    int ph = bid & 31;
    int c  = (bid >> 5) & 1023;
    int b  = bid >> 15;
    int t  = threadIdx.x;                       // == w (0..255)

    const float* xp = x + (((size_t)(b * C_ + c) * H_) + (size_t)ph * 8) * W_ + t;
    float s = 0.f;
#pragma unroll
    for (int r = 0; r < 8; ++r) s += xp[(size_t)r * W_];

    // reduce the 8 lanes that share one patch (lane == w here)
    s += __shfl_xor(s, 1, 32);
    s += __shfl_xor(s, 2, 32);
    s += __shfl_xor(s, 4, 32);

    if ((t & 7) == 0) {
        int pw = t >> 3;
        int p  = ph * 32 + pw;
        yt[((size_t)b * PPAD + (p + 1)) * C_ + c] = (_Float16)(s * (1.0f / 64.0f));
    }
    // zero the padding rows (p = -1 and p = 1024), once per (b,c)
    if (t == 0 && ph == 0)  yt[((size_t)b * PPAD + 0)    * C_ + c] = (_Float16)0.f;
    if (t == 0 && ph == 31) yt[((size_t)b * PPAD + 1025) * C_ + c] = (_Float16)0.f;
}

// ---------------------------------------------------------------------------
// Kernel 3: conv1d-as-GEMM with v_wmma_f32_16x16x32_f16, fused sigmoid.
// One wave32 computes one 16(co) x 16(p) tile of g for one batch.
// 3 taps -> 3 independent accumulators (breaks WMMA RAW chains).
// ---------------------------------------------------------------------------
__global__ __launch_bounds__(256)
void gemm_kernel(const _Float16* __restrict__ yt,
                 const _Float16* __restrict__ wt,
                 float* __restrict__ g) {
    int wave = threadIdx.y;
    int tile = blockIdx.x * 8 + wave;           // 0..8191
    int co_t = tile & 63;                       // 64 co tiles of 16
    int npt  = tile >> 6;                       // 0..127
    int b    = npt >> 6;
    int p_t  = npt & 63;                        // 64 p tiles of 16

    int lane = threadIdx.x;                     // 0..31
    int half = lane >> 4;
    int l16  = lane & 15;

    // A: wt[k][co_t*16 + l16][ci0 + half*8 (+0 / +16)]
    const _Float16* a0 = wt + (size_t)(co_t * 16 + l16) * C_ + half * 8;
    // B: yt[b][p_t*16 + l16 + k][ci0 + half*8 (+0 / +16)]
    const _Float16* b0 = yt + (size_t)b * PPAD * C_ + half * 8;

    v8f acc0 = {}, acc1 = {}, acc2 = {};

    for (int ci0 = 0; ci0 < C_; ci0 += 32) {
#pragma unroll
        for (int k = 0; k < 3; ++k) {
            const _Float16* ap = a0 + (size_t)k * C_ * C_ + ci0;
            v8h al = *(const v8h*)(ap);
            v8h ah = *(const v8h*)(ap + 16);
            v16h A = __builtin_shufflevector(al, ah,
                0, 1, 2, 3, 4, 5, 6, 7, 8, 9, 10, 11, 12, 13, 14, 15);

            const _Float16* bp = b0 + (size_t)(p_t * 16 + l16 + k) * C_ + ci0;
            v8h bl = *(const v8h*)(bp);
            v8h bh = *(const v8h*)(bp + 16);
            v16h Bm = __builtin_shufflevector(bl, bh,
                0, 1, 2, 3, 4, 5, 6, 7, 8, 9, 10, 11, 12, 13, 14, 15);

            if (k == 0)
                acc0 = __builtin_amdgcn_wmma_f32_16x16x32_f16(false, A, false, Bm,
                                                              (short)0, acc0, false, false);
            else if (k == 1)
                acc1 = __builtin_amdgcn_wmma_f32_16x16x32_f16(false, A, false, Bm,
                                                              (short)0, acc1, false, false);
            else
                acc2 = __builtin_amdgcn_wmma_f32_16x16x32_f16(false, A, false, Bm,
                                                              (short)0, acc2, false, false);
        }
    }

    // D layout: VGPR i -> M = i + 8*half, lane -> N = l16
    float* gp = g + ((size_t)b * C_ + (size_t)co_t * 16) * P_ + p_t * 16 + l16;
#pragma unroll
    for (int i = 0; i < 8; ++i) {
        float v = acc0[i] + acc1[i] + acc2[i];
        float s = 1.0f / (1.0f + __expf(-v));
        gp[(size_t)(i + half * 8) * P_] = s;
    }
}

// ---------------------------------------------------------------------------
// Kernel 4: out = x * gate, float4-vectorized streaming.
// Each float4 (4 consecutive w) lies inside one 8-wide patch.
// ---------------------------------------------------------------------------
__global__ void apply_kernel(const float* __restrict__ x,
                             const float* __restrict__ g,
                             float* __restrict__ out) {
    size_t i4 = (size_t)blockIdx.x * blockDim.x + threadIdx.x;  // index in float4 units
    int    w4 = (int)(i4 & 63);          // w / 4
    int    h  = (int)((i4 >> 6) & 255);
    size_t bc = i4 >> 14;                // b*C + c
    float gv = g[bc * P_ + (size_t)(h >> 3) * WF_ + (w4 >> 1)];
    v4f xv = ((const v4f*)x)[i4];
    ((v4f*)out)[i4] = xv * gv;
}

// ---------------------------------------------------------------------------
extern "C" void kernel_launch(void* const* d_in, const int* in_sizes, int n_in,
                              void* d_out, int out_size, void* d_ws, size_t ws_size,
                              hipStream_t stream) {
    const float* x  = (const float*)d_in[0];   // [2,1024,256,256] f32
    const float* cw = (const float*)d_in[1];   // [1024,1024,3]    f32
    float* out = (float*)d_out;                // [2,1024,256,256] f32

    char* ws = (char*)d_ws;
    float*    g  = (float*)(ws + G_OFF);
    _Float16* yt = (_Float16*)(ws + YT_OFF);
    _Float16* wt = (_Float16*)(ws + WT_OFF);

    // 1) weight convert/transpose: [co][ci][3] f32 -> [3][co][ci] f16
    wcvt_kernel<<<(C_ * C_) / 256, 256, 0, stream>>>(cw, wt);

    // 2) patch mean pool -> yt[b][p+1][ci] f16 (with zero pad rows)
    pool_kernel<<<B_ * C_ * HF_, 256, 0, stream>>>(x, yt);

    // 3) conv1d as f16 WMMA GEMM + sigmoid -> g[b][c][p] f32
    dim3 bb(32, 8, 1);
    gemm_kernel<<<(64 * 128) / 8, bb, 0, stream>>>(yt, wt, g);

    // 4) gate broadcast multiply (streaming, float4)
    apply_kernel<<<(B_ * C_ * H_ * (W_ / 4)) / 256, 256, 0, stream>>>(x, g, out);
}
